// GatedAxialAttention_66005057405744
// MI455X (gfx1250) — matrix operand order repaired and checked by reading, not compiled, not run on hardware
//
#include <hip/hip_runtime.h>
#include <math.h>

#define DIMC 512
#define NH   8
#define HD   64
#define BATCH 4
#define HH   128
#define WW   128
#define BH_  (BATCH*HH)      // 512 sequences
#define TOK  (BH_*WW)        // 65536 tokens

typedef _Float16 h16;
typedef __attribute__((ext_vector_type(16))) _Float16 v16h;
typedef __attribute__((ext_vector_type(8)))  _Float16 v8h;
typedef __attribute__((ext_vector_type(8)))  float    v8f;
typedef __attribute__((ext_vector_type(4)))  unsigned u32x4;
typedef __attribute__((ext_vector_type(8)))  unsigned u32x8;

union F16x16 { v16h v; v8h h[2]; };

__device__ __forceinline__ v8f wmma_f16(v16h a, v16h b, v8f c) {
  return __builtin_amdgcn_wmma_f32_16x16x32_f16(false, a, false, b, (short)0, c,
                                                false, false);
}

// ---------------------------------------------------------------------------
// Tensor Data Mover: DMA one 2D tile (tile_rows x 32 halves, row stride 512
// halves) from global to LDS.  D# per CDNA5 ISA ch.8: group0 = count/lds/
// global addr/type, group1 = data_size + tensor/tile dims + strides.
// Issued by one wave; tracked with TENSORcnt.
// ---------------------------------------------------------------------------
__device__ __forceinline__ void tdm_load_tile_2d(unsigned lds_addr,
                                                 const void* gtile) {
  const unsigned long long ga = (unsigned long long)gtile;
  u32x4 g0;
  g0[0] = 1u;                                       // count=1, user mode
  g0[1] = lds_addr;                                 // LDS byte address
  g0[2] = (unsigned)ga;                             // global_addr[31:0]
  g0[3] = (unsigned)((ga >> 32) & 0x01FFFFFFu)      // global_addr[56:32]
          | (2u << 30);                             // type=2 (image)
  u32x8 g1;
  g1[0] = 1u << 16;                                 // data_size=1 (2 bytes)
  g1[1] = (512u & 0xFFFFu) << 16;                   // tensor_dim0[15:0]=512
  g1[2] = (512u >> 16) | ((1u << 20 & 0xFFFFu) << 16); // dim0 hi / dim1 lo
  g1[3] = ((1u << 20) >> 16) | (32u << 16);         // dim1 hi / tile_dim0=32
  g1[4] = 128u;                                     // tile_dim1=128, tile_dim2=0
  g1[5] = 512u;                                     // tensor_dim0_stride lo32
  g1[6] = 0u;                                       // stride hi / dim1_stride lo
  g1[7] = 0u;
  asm volatile("tensor_load_to_lds %0, %1" :: "s"(g0), "s"(g1) : "memory");
}
__device__ __forceinline__ void wait_tensor0() {
  __builtin_amdgcn_s_wait_tensorcnt(0);
}

// ---------------------------------------------------------------------------
// Kernel 0: transpose-pack x(B,C,H,W) f32 -> xf(token, C) f16
// ---------------------------------------------------------------------------
__global__ void k_pack_x(const float* __restrict__ x, h16* __restrict__ xf) {
  __shared__ float tile[32][33];
  const int w0 = blockIdx.x * 32;
  const int c0 = blockIdx.y * 32;
  const int bh = blockIdx.z;
  const int b = bh >> 7, h = bh & 127;
  const int tx = threadIdx.x, ty = threadIdx.y;
#pragma unroll
  for (int i = 0; i < 4; ++i) {
    const int c = c0 + ty + i * 8;
    tile[ty + i * 8][tx] = x[(((size_t)b * DIMC + c) * HH + h) * WW + w0 + tx];
  }
  __syncthreads();
#pragma unroll
  for (int i = 0; i < 4; ++i) {
    const int w = w0 + ty + i * 8;
    xf[(size_t)(bh * WW + w) * DIMC + c0 + tx] = (h16)tile[tx][ty + i * 8];
  }
}

// ---------------------------------------------------------------------------
// Kernel 1: generic f32 -> f16 convert
// ---------------------------------------------------------------------------
__global__ void k_cvt(const float* __restrict__ s, h16* __restrict__ d, int n) {
  int i = blockIdx.x * blockDim.x + threadIdx.x;
  if (i < n) d[i] = (h16)s[i];
}

// ---------------------------------------------------------------------------
// Kernel 2: fused QKV GEMM.  A = xf[TOK][512], B = wqkv[1536][512] (row = out
// channel, K contiguous).  BM=BN=128, BK=32, 8 waves, wave tile 32x64.
// Tiles staged to LDS by the Tensor Data Mover, double buffered.
// Epilogue: +bias, scatter to q/k/v as (bh, head, w, hd) f16.
// ---------------------------------------------------------------------------
__global__ __launch_bounds__(256) void k_qkv(
    const h16* __restrict__ A, const h16* __restrict__ Bm,
    const float* __restrict__ qb, const float* __restrict__ kb,
    const float* __restrict__ vb,
    h16* __restrict__ qo, h16* __restrict__ ko, h16* __restrict__ vo) {
  __shared__ __align__(16) h16 As[2][128 * 32];
  __shared__ __align__(16) h16 Bs[2][128 * 32];
  const int tid = threadIdx.x, lane = tid & 31, wid = tid >> 5;
  const int bm = blockIdx.x, bn = blockIdx.y;
  const long arow = (long)bm * 128;
  const long brow = (long)bn * 128;
  const int m0 = (wid & 3) * 32, n0 = (wid >> 2) * 64;

  v8f acc[2][4];
  const v8f zro = {0.f, 0.f, 0.f, 0.f, 0.f, 0.f, 0.f, 0.f};
#pragma unroll
  for (int i = 0; i < 2; ++i)
#pragma unroll
    for (int j = 0; j < 4; ++j) acc[i][j] = zro;

  if (wid == 0) {
    tdm_load_tile_2d((unsigned)(unsigned long long)&As[0][0],
                     A + arow * DIMC);
    tdm_load_tile_2d((unsigned)(unsigned long long)&Bs[0][0],
                     Bm + brow * DIMC);
  }
  int buf = 0;
  for (int kt = 0; kt < 16; ++kt) {
    if (wid == 0) wait_tensor0();
    __syncthreads();
    if (wid == 0 && kt + 1 < 16) {
      const int kk = (kt + 1) * 32;
      tdm_load_tile_2d((unsigned)(unsigned long long)&As[buf ^ 1][0],
                       A + arow * DIMC + kk);
      tdm_load_tile_2d((unsigned)(unsigned long long)&Bs[buf ^ 1][0],
                       Bm + brow * DIMC + kk);
    }
    F16x16 af[2], bf[4];
    const int aoff = (lane >> 4) * 8;
    const int boff = (lane >> 4) * 8;
#pragma unroll
    for (int mt = 0; mt < 2; ++mt) {
      const int r = m0 + mt * 16 + (lane & 15);
      af[mt].h[0] = *(const v8h*)&As[buf][r * 32 + aoff];
      af[mt].h[1] = *(const v8h*)&As[buf][r * 32 + aoff + 16];
    }
#pragma unroll
    for (int nt = 0; nt < 4; ++nt) {
      const int r = n0 + nt * 16 + (lane & 15);
      bf[nt].h[0] = *(const v8h*)&Bs[buf][r * 32 + boff];
      bf[nt].h[1] = *(const v8h*)&Bs[buf][r * 32 + boff + 16];
    }
#pragma unroll
    for (int mt = 0; mt < 2; ++mt)
#pragma unroll
      for (int nt = 0; nt < 4; ++nt)
        acc[mt][nt] = wmma_f16(af[mt].v, bf[nt].v, acc[mt][nt]);
    buf ^= 1;
  }
  // Epilogue: M-tile == exactly one (b,h) row of 128 tokens (w = m local).
  const int bhh = bm;
#pragma unroll
  for (int mt = 0; mt < 2; ++mt)
#pragma unroll
    for (int nt = 0; nt < 4; ++nt) {
      const int ng = bn * 128 + n0 + nt * 16 + (lane & 15);  // 0..1535
      const int sel = ng >> 9;
      const int nn = ng & 511;
      const int head = nn >> 6, hd = nn & 63;
      const float* bias = sel == 0 ? qb : (sel == 1 ? kb : vb);
      h16* dst = sel == 0 ? qo : (sel == 1 ? ko : vo);
#pragma unroll
      for (int r = 0; r < 8; ++r) {
        const int ml = m0 + mt * 16 + r + 8 * (lane >> 4);  // w
        const float val = acc[mt][nt][r] + bias[nn];
        dst[(((size_t)bhh * NH + head) * WW + ml) * HD + hd] = (h16)val;
      }
    }
}

// ---------------------------------------------------------------------------
// Kernel 3: attention, one block per (bh, head).
// S = q @ (k + gq*pq)^T + (gk*pk) @ k^T ; softmax(S*0.125) ; O = P @ vmix.
// ---------------------------------------------------------------------------
__global__ __launch_bounds__(256) void k_attn(
    const h16* __restrict__ q, const h16* __restrict__ k,
    const h16* __restrict__ v, const h16* __restrict__ pq,
    const h16* __restrict__ pk, const h16* __restrict__ pv,
    const float* __restrict__ gq_p, const float* __restrict__ gk_p,
    const float* __restrict__ gv1_p, const float* __restrict__ gv2_p,
    h16* __restrict__ y) {
  __shared__ __align__(128) char smem[81920];
  h16* q_s   = (h16*)(smem);            // [128][64]
  h16* kk_s  = (h16*)(smem + 16384);    // k + gq*pq
  h16* pkg_s = (h16*)(smem + 32768);    // gk*pk
  h16* k_s   = (h16*)(smem + 49152);
  h16* vmT_s = (h16*)(smem + 65536);    // [64][128] transposed vmix
  h16* P_s   = (h16*)(smem);            // aliases q_s/kk_s after S is done

  const int bh = blockIdx.x, head = blockIdx.y;
  const int tid = threadIdx.x, lane = tid & 31, wid = tid >> 5;
  const float gq = *gq_p, gk = *gk_p, gv1 = *gv1_p, gv2 = *gv2_p;

  const h16* qp  = q + ((size_t)bh * NH + head) * WW * HD;
  const h16* kp  = k + ((size_t)bh * NH + head) * WW * HD;
  const h16* vp  = v + ((size_t)bh * NH + head) * WW * HD;
  const h16* pqp = pq + (size_t)head * WW * HD;
  const h16* pkp = pk + (size_t)head * WW * HD;
  const h16* pvp = pv + (size_t)head * WW * HD;

  for (int base = tid * 8; base < WW * HD; base += 256 * 8) {
    const v8h qv = *(const v8h*)(qp + base);
    const v8h kv = *(const v8h*)(kp + base);
    const v8h vv = *(const v8h*)(vp + base);
    const v8h pqv = *(const v8h*)(pqp + base);
    const v8h pkv = *(const v8h*)(pkp + base);
    const v8h pvv = *(const v8h*)(pvp + base);
    const int j = base >> 6;   // row (w)
    const int d0 = base & 63;  // hd
    *(v8h*)(q_s + base) = qv;
    *(v8h*)(k_s + base) = kv;
    v8h kkv, pkgv;
#pragma unroll
    for (int e = 0; e < 8; ++e) {
      kkv[e]  = (h16)((float)kv[e] + gq * (float)pqv[e]);
      pkgv[e] = (h16)(gk * (float)pkv[e]);
      vmT_s[(d0 + e) * WW + j] = (h16)(gv1 * (float)vv[e] + gv2 * (float)pvv[e]);
    }
    *(v8h*)(kk_s + base) = kkv;
    *(v8h*)(pkg_s + base) = pkgv;
  }
  __syncthreads();

  v8f sacc[8];
  const v8f zro = {0.f, 0.f, 0.f, 0.f, 0.f, 0.f, 0.f, 0.f};
#pragma unroll
  for (int i = 0; i < 8; ++i) sacc[i] = zro;

#pragma unroll
  for (int pass = 0; pass < 2; ++pass) {
    const h16* Aop = pass ? pkg_s : q_s;
    const h16* Bop = pass ? k_s : kk_s;
#pragma unroll
    for (int ks = 0; ks < 2; ++ks) {
      F16x16 af;
      const int ar = wid * 16 + (lane & 15);
      const int ao = ks * 32 + (lane >> 4) * 8;
      af.h[0] = *(const v8h*)&Aop[ar * HD + ao];
      af.h[1] = *(const v8h*)&Aop[ar * HD + ao + 16];
#pragma unroll
      for (int nt = 0; nt < 8; ++nt) {
        F16x16 bfr;
        const int br = nt * 16 + (lane & 15);
        const int bo = ks * 32 + (lane >> 4) * 8;
        bfr.h[0] = *(const v8h*)&Bop[br * HD + bo];
        bfr.h[1] = *(const v8h*)&Bop[br * HD + bo + 16];
        sacc[nt] = wmma_f16(af.v, bfr.v, sacc[nt]);
      }
    }
  }
  __syncthreads();  // operand buffers dead; safe to alias P_s

  // Softmax over each 128-wide row; rows r / r+8 live in lane halves.
  const float scale = 0.125f;  // 64^-0.5
#pragma unroll
  for (int r = 0; r < 8; ++r) {
    float mx = -3.0e38f;
#pragma unroll
    for (int nt = 0; nt < 8; ++nt) mx = fmaxf(mx, sacc[nt][r]);
#pragma unroll
    for (int m = 1; m <= 8; m <<= 1) mx = fmaxf(mx, __shfl_xor(mx, m, 32));
    float sum = 0.f;
#pragma unroll
    for (int nt = 0; nt < 8; ++nt) {
      const float p = __expf(scale * (sacc[nt][r] - mx));
      sacc[nt][r] = p;
      sum += p;
    }
#pragma unroll
    for (int m = 1; m <= 8; m <<= 1) sum += __shfl_xor(sum, m, 32);
    const float inv = 1.0f / sum;
    const int row = wid * 16 + r + 8 * (lane >> 4);
#pragma unroll
    for (int nt = 0; nt < 8; ++nt)
      P_s[row * 128 + nt * 16 + (lane & 15)] = (h16)(sacc[nt][r] * inv);
  }
  __syncthreads();

  v8f oacc[4];
#pragma unroll
  for (int i = 0; i < 4; ++i) oacc[i] = zro;
#pragma unroll
  for (int ks = 0; ks < 4; ++ks) {
    F16x16 af;
    const int ar = wid * 16 + (lane & 15);
    const int ao = ks * 32 + (lane >> 4) * 8;
    af.h[0] = *(const v8h*)&P_s[ar * 128 + ao];
    af.h[1] = *(const v8h*)&P_s[ar * 128 + ao + 16];
#pragma unroll
    for (int nt = 0; nt < 4; ++nt) {
      F16x16 bfr;
      const int br = nt * 16 + (lane & 15);       // hd
      const int bo = ks * 32 + (lane >> 4) * 8;  // j offset
      bfr.h[0] = *(const v8h*)&vmT_s[br * 128 + bo];
      bfr.h[1] = *(const v8h*)&vmT_s[br * 128 + bo + 16];
      oacc[nt] = wmma_f16(af.v, bfr.v, oacc[nt]);
    }
  }
#pragma unroll
  for (int nt = 0; nt < 4; ++nt)
#pragma unroll
    for (int r = 0; r < 8; ++r) {
      const int m = wid * 16 + r + 8 * (lane >> 4);
      const int c = head * HD + nt * 16 + (lane & 15);
      y[((size_t)bh * WW + m) * DIMC + c] = (h16)oacc[nt][r];
    }
}

// ---------------------------------------------------------------------------
// Kernel 4: output projection + transpose store to (B, C, H, W) f32.
// ---------------------------------------------------------------------------
__global__ __launch_bounds__(256) void k_oproj(const h16* __restrict__ A,
                                               const h16* __restrict__ Bm,
                                               const float* __restrict__ ob,
                                               float* __restrict__ out) {
  __shared__ __align__(16) char smem[66048];
  h16* Asb = (h16*)smem;             // 2 x 4096 halves
  h16* Bsb = (h16*)(smem + 16384);   // 2 x 4096 halves
  float* tile = (float*)smem;        // [128][129] f32, reused after mainloop

  const int tid = threadIdx.x, lane = tid & 31, wid = tid >> 5;
  const int bm = blockIdx.x, bn = blockIdx.y;
  const long arow = (long)bm * 128;
  const long brow = (long)bn * 128;
  const int m0 = (wid & 3) * 32, n0 = (wid >> 2) * 64;

  v8f acc[2][4];
  const v8f zro = {0.f, 0.f, 0.f, 0.f, 0.f, 0.f, 0.f, 0.f};
#pragma unroll
  for (int i = 0; i < 2; ++i)
#pragma unroll
    for (int j = 0; j < 4; ++j) acc[i][j] = zro;

  if (wid == 0) {
    tdm_load_tile_2d((unsigned)(unsigned long long)&Asb[0], A + arow * DIMC);
    tdm_load_tile_2d((unsigned)(unsigned long long)&Bsb[0], Bm + brow * DIMC);
  }
  int buf = 0;
  for (int kt = 0; kt < 16; ++kt) {
    if (wid == 0) wait_tensor0();
    __syncthreads();
    if (wid == 0 && kt + 1 < 16) {
      const int kk = (kt + 1) * 32;
      tdm_load_tile_2d((unsigned)(unsigned long long)&Asb[(buf ^ 1) * 4096],
                       A + arow * DIMC + kk);
      tdm_load_tile_2d((unsigned)(unsigned long long)&Bsb[(buf ^ 1) * 4096],
                       Bm + brow * DIMC + kk);
    }
    F16x16 af[2], bf[4];
    const int aoff = (lane >> 4) * 8;
    const int boff = (lane >> 4) * 8;
#pragma unroll
    for (int mt = 0; mt < 2; ++mt) {
      const int r = m0 + mt * 16 + (lane & 15);
      af[mt].h[0] = *(const v8h*)&Asb[buf * 4096 + r * 32 + aoff];
      af[mt].h[1] = *(const v8h*)&Asb[buf * 4096 + r * 32 + aoff + 16];
    }
#pragma unroll
    for (int nt = 0; nt < 4; ++nt) {
      const int r = n0 + nt * 16 + (lane & 15);
      bf[nt].h[0] = *(const v8h*)&Bsb[buf * 4096 + r * 32 + boff];
      bf[nt].h[1] = *(const v8h*)&Bsb[buf * 4096 + r * 32 + boff + 16];
    }
#pragma unroll
    for (int mt = 0; mt < 2; ++mt)
#pragma unroll
      for (int nt = 0; nt < 4; ++nt)
        acc[mt][nt] = wmma_f16(af[mt].v, bf[nt].v, acc[mt][nt]);
    buf ^= 1;
  }
  __syncthreads();  // done with Asb/Bsb, reuse as f32 tile
#pragma unroll
  for (int mt = 0; mt < 2; ++mt)
#pragma unroll
    for (int nt = 0; nt < 4; ++nt) {
      const int nl = n0 + nt * 16 + (lane & 15);
      const float bv = ob[bn * 128 + nl];
#pragma unroll
      for (int r = 0; r < 8; ++r) {
        const int ml = m0 + mt * 16 + r + 8 * (lane >> 4);
        tile[ml * 129 + nl] = acc[mt][nt][r] + bv;
      }
    }
  __syncthreads();
  const int b = bm >> 7, h = bm & 127;
  for (int idx = tid; idx < 128 * 128; idx += 256) {
    const int cl = idx >> 7, w = idx & 127;
    out[(((size_t)b * DIMC + bn * 128 + cl) * HH + h) * WW + w] =
        tile[w * 129 + cl];
  }
}

// ---------------------------------------------------------------------------
extern "C" void kernel_launch(void* const* d_in, const int* in_sizes, int n_in,
                              void* d_out, int out_size, void* d_ws,
                              size_t ws_size, hipStream_t stream) {
  const float* x    = (const float*)d_in[0];
  const float* q_w  = (const float*)d_in[1];
  const float* q_b  = (const float*)d_in[2];
  const float* k_w  = (const float*)d_in[3];
  const float* k_b  = (const float*)d_in[4];
  const float* v_w  = (const float*)d_in[5];
  const float* v_b  = (const float*)d_in[6];
  const float* o_w  = (const float*)d_in[7];
  const float* o_b  = (const float*)d_in[8];
  const float* pos_q = (const float*)d_in[9];
  const float* pos_k = (const float*)d_in[10];
  const float* pos_v = (const float*)d_in[11];
  const float* g_q  = (const float*)d_in[12];
  const float* g_k  = (const float*)d_in[13];
  const float* g_v1 = (const float*)d_in[14];
  const float* g_v2 = (const float*)d_in[15];

  char* ws = (char*)d_ws;
  size_t off = 0;
  const size_t tokC = (size_t)TOK * DIMC * sizeof(h16);  // 64 MB
  h16* xf   = (h16*)(ws + off); off += tokC;
  h16* wqkv = (h16*)(ws + off); off += (size_t)3 * DIMC * DIMC * sizeof(h16);
  h16* wo   = (h16*)(ws + off); off += (size_t)DIMC * DIMC * sizeof(h16);
  h16* pqh  = (h16*)(ws + off); off += (size_t)NH * WW * HD * sizeof(h16);
  h16* pkh  = (h16*)(ws + off); off += (size_t)NH * WW * HD * sizeof(h16);
  h16* pvh  = (h16*)(ws + off); off += (size_t)NH * WW * HD * sizeof(h16);
  h16* qh   = (h16*)(ws + off); off += tokC;
  h16* kh   = (h16*)(ws + off); off += tokC;
  h16* vh   = (h16*)(ws + off); off += tokC;
  h16* yh   = (h16*)(ws + off); off += tokC;

  k_pack_x<<<dim3(WW / 32, DIMC / 32, BH_), dim3(32, 8), 0, stream>>>(x, xf);

  const int nw = DIMC * DIMC;
  k_cvt<<<(nw + 255) / 256, 256, 0, stream>>>(q_w, wqkv, nw);
  k_cvt<<<(nw + 255) / 256, 256, 0, stream>>>(k_w, wqkv + nw, nw);
  k_cvt<<<(nw + 255) / 256, 256, 0, stream>>>(v_w, wqkv + 2 * nw, nw);
  k_cvt<<<(nw + 255) / 256, 256, 0, stream>>>(o_w, wo, nw);
  const int np = NH * WW * HD;
  k_cvt<<<(np + 255) / 256, 256, 0, stream>>>(pos_q, pqh, np);
  k_cvt<<<(np + 255) / 256, 256, 0, stream>>>(pos_k, pkh, np);
  k_cvt<<<(np + 255) / 256, 256, 0, stream>>>(pos_v, pvh, np);

  k_qkv<<<dim3(TOK / 128, (3 * DIMC) / 128), 256, 0, stream>>>(
      xf, wqkv, q_b, k_b, v_b, qh, kh, vh);

  k_attn<<<dim3(BH_, NH), 256, 0, stream>>>(qh, kh, vh, pqh, pkh, pvh, g_q,
                                            g_k, g_v1, g_v2, yh);

  k_oproj<<<dim3(TOK / 128, DIMC / 128), 256, 0, stream>>>(yh, wo, o_b,
                                                           (float*)d_out);
}
